// AttentionOnly_60249801228687
// MI455X (gfx1250) — matrix operand order repaired and checked by reading, not compiled, not run on hardware
//
#include <hip/hip_runtime.h>
#include <hip/hip_bf16.h>

// ---------------------------------------------------------------------------
// GATv2 (3 layers) + graph-max readout + MLP classifier for MI455X (gfx1250).
// GEMMs: v_wmma_f32_16x16x32_f16, LDS double-buffered via
// global_load_async_to_lds_b128 (ASYNCcnt) with pre-converted f16 operands.
// ---------------------------------------------------------------------------

typedef __attribute__((ext_vector_type(16))) _Float16 v16h;
typedef __attribute__((ext_vector_type(8)))  float    v8f;

#define N_NODES  50000
#define E_EDGES  800000
#define G_GRAPHS 64
#define IN_DIM   128
#define HID      256
#define HEADS    8
#define DHEAD    32
#define OUT_DIM  10
#define NEG_SLOPE 0.2f

// ---- WMMA GEMM tile config -------------------------------------------------
#define BM 128
#define BN 64
#define BK 32
#define LDA 40   // BK + 8 halfs pad
#define LDB 40   // BK + 8 halfs pad (B tile stored [col][k])

union Frag16 { v16h h; unsigned int u[8]; };

__device__ __forceinline__ unsigned lds_addr_u32(const void* p) {
    // addrspace(3) offset == low 32 bits of the generic pointer (ISA 10.2)
    return (unsigned)(unsigned long long)p;
}

// C[M,N] = A16[M,K] x BT16[N,K]^T + bias, optional ReLU, optional f16 copy.
// Tiles staged into double-buffered LDS with async-to-LDS copies.
__global__ __launch_bounds__(256)
void gemm_bias_wmma(const _Float16* __restrict__ A16, const _Float16* __restrict__ BT16,
                    const float* __restrict__ bias, float* __restrict__ C,
                    _Float16* __restrict__ C16, int M, int N, int K, int relu)
{
    __shared__ _Float16 As[2][BM * LDA];
    __shared__ _Float16 Bt[2][BN * LDB];

    const int tid  = threadIdx.x;
    const int lane = tid & 31;
    const int wave = tid >> 5;
    const int wm   = (wave & 3) * 32;
    const int wn   = (wave >> 2) * 32;
    const int half = lane >> 4;
    const int ln   = lane & 15;
    const int row0 = blockIdx.y * BM;
    const int col0 = blockIdx.x * BN;

    // ---- per-lane async-copy addressing (16B chunks, uniform exec) ----
    // A tile: 128 rows x 32 halfs = 512 chunks -> 2 instr/wave
    // B tile:  64 cols x 32 halfs = 256 chunks -> 1 instr/wave
    const int idA0 = wave * 64 + lane;
    const int idA1 = idA0 + 32;
    const int idB  = wave * 32 + lane;
    const int rA0 = idA0 >> 2, cA0 = (idA0 & 3) * 8;
    const int rA1 = idA1 >> 2, cA1 = (idA1 & 3) * 8;
    const int cB  = idB  >> 2, kB  = (idB  & 3) * 8;
    int grA0 = row0 + rA0; if (grA0 >= M) grA0 = M - 1;   // clamp: OOB rows only
    int grA1 = row0 + rA1; if (grA1 >= M) grA1 = M - 1;   // feed store-guarded C
    int gcB  = col0 + cB;  if (gcB  >= N) gcB  = N - 1;
    const unsigned gA0 = (unsigned)(grA0 * K + cA0) * 2u;
    const unsigned gA1 = (unsigned)(grA1 * K + cA1) * 2u;
    const unsigned gB0 = (unsigned)(gcB  * K + kB ) * 2u;
    const unsigned ldsA0 = lds_addr_u32(&As[0][rA0 * LDA + cA0]);
    const unsigned ldsA1 = lds_addr_u32(&As[0][rA1 * LDA + cA1]);
    const unsigned ldsB0 = lds_addr_u32(&Bt[0][cB  * LDB + kB ]);
    const unsigned bufStrideA = (unsigned)(BM * LDA * 2);
    const unsigned bufStrideB = (unsigned)(BN * LDB * 2);
    const unsigned long long baseA = (unsigned long long)A16;
    const unsigned long long baseB = (unsigned long long)BT16;

    auto issue_tile = [&](int kt, int buf) {
        const unsigned koff = (unsigned)kt * (BK * 2);
        const unsigned bA = buf ? bufStrideA : 0u;
        const unsigned bB = buf ? bufStrideB : 0u;
        asm volatile("global_load_async_to_lds_b128 %0, %1, %2"
                     :: "v"(ldsA0 + bA), "v"(gA0 + koff), "s"(baseA) : "memory");
        asm volatile("global_load_async_to_lds_b128 %0, %1, %2"
                     :: "v"(ldsA1 + bA), "v"(gA1 + koff), "s"(baseA) : "memory");
        asm volatile("global_load_async_to_lds_b128 %0, %1, %2"
                     :: "v"(ldsB0 + bB), "v"(gB0 + koff), "s"(baseB) : "memory");
    };

    v8f acc[2][2] = {};
    const int nk = K / BK;

    issue_tile(0, 0);                       // prologue prefetch

    for (int kt = 0; kt < nk; ++kt) {
        const int buf = kt & 1;
        if (kt + 1 < nk) {
            issue_tile(kt + 1, buf ^ 1);    // prefetch next tile
            asm volatile("s_wait_asynccnt 0x3" ::: "memory"); // tile kt resident
        } else {
            asm volatile("s_wait_asynccnt 0x0" ::: "memory");
        }
        __syncthreads();                    // all waves' copies visible

        Frag16 fa[2], fb[2];
        #pragma unroll
        for (int t = 0; t < 2; ++t) {
            // A 16x32 f16: lane = row M=ln; element i -> K = half*16 + i
            const unsigned int* arow =
                reinterpret_cast<const unsigned int*>(&As[buf][(wm + t * 16 + ln) * LDA]);
            #pragma unroll
            for (int j = 0; j < 4; ++j) {
                fa[t].u[j]     = arow[half * 8 + j];
                fa[t].u[4 + j] = arow[half * 8 + 4 + j];
            }
            // B 32x16 f16: lane = col N=ln; element i -> K = half*16 + i
            const unsigned int* brow =
                reinterpret_cast<const unsigned int*>(&Bt[buf][(wn + t * 16 + ln) * LDB]);
            #pragma unroll
            for (int j = 0; j < 8; ++j)
                fb[t].u[j] = brow[half * 8 + j];
        }

        #pragma unroll
        for (int ta = 0; ta < 2; ++ta)
            #pragma unroll
            for (int tb = 0; tb < 2; ++tb)
                acc[ta][tb] = __builtin_amdgcn_wmma_f32_16x16x32_f16(
                    false, fa[ta].h, false, fb[tb].h,
                    (short)0, acc[ta][tb], false, false);

        __syncthreads();                    // protect buf before next issue refills it
    }

    // ---- store C (16x16 f32 layout: VGPR j -> M = j + 8*half, N = ln) ----
    #pragma unroll
    for (int ta = 0; ta < 2; ++ta) {
        #pragma unroll
        for (int tb = 0; tb < 2; ++tb) {
            int col = col0 + wn + tb * 16 + ln;
            if (col >= N) continue;
            float bv = bias[col];
            int rbase = row0 + wm + ta * 16 + half * 8;
            #pragma unroll
            for (int j = 0; j < 8; ++j) {
                int r = rbase + j;
                if (r < M) {
                    float v = acc[ta][tb][j] + bv;
                    if (relu && v < 0.f) v = 0.f;
                    C[(size_t)r * N + col] = v;
                    if (C16) C16[(size_t)r * N + col] = (_Float16)v;
                }
            }
        }
    }
}

// ---- conversion kernels ----------------------------------------------------
__global__ __launch_bounds__(256)
void cvt_f32_f16(const float* __restrict__ in, _Float16* __restrict__ out, int n)
{
    int t = blockIdx.x * blockDim.x + threadIdx.x;
    if (t < n) out[t] = (_Float16)in[t];
}

// out[c*R + r] = in[r*C + c]   (B [K,N] f32 -> BT [N,K] f16)
__global__ __launch_bounds__(256)
void cvt_transpose_f32_f16(const float* __restrict__ in, _Float16* __restrict__ out,
                           int R, int C)
{
    int t = blockIdx.x * blockDim.x + threadIdx.x;
    if (t >= R * C) return;
    int r = t / C, c = t % C;
    out[(size_t)c * R + r] = (_Float16)in[t];
}

// ---- monotonic float <-> uint mapping for atomic max on floats -------------
__device__ __forceinline__ unsigned int fmap(float f) {
    unsigned int u = __float_as_uint(f);
    return (u & 0x80000000u) ? ~u : (u | 0x80000000u);
}
__device__ __forceinline__ float funmap(unsigned int k) {
    unsigned int u = (k & 0x80000000u) ? (k & 0x7fffffffu) : ~k;
    return __uint_as_float(u);
}

// ---- per-(edge, head) GATv2 score: a^T leaky_relu(hs[src] + hd[dst]) -------
__global__ __launch_bounds__(256)
void edge_scores(const float* __restrict__ hs, const float* __restrict__ hd,
                 const float* __restrict__ attn, const int* __restrict__ src,
                 const int* __restrict__ dst, float* __restrict__ e)
{
    int t = blockIdx.x * blockDim.x + threadIdx.x;
    if (t >= E_EDGES * HEADS) return;
    int eidx = t >> 3, h = t & 7;
    int s = src[eidx], d = dst[eidx];
    const float4* ps = reinterpret_cast<const float4*>(hs + (size_t)s * HID + h * DHEAD);
    const float4* pd = reinterpret_cast<const float4*>(hd + (size_t)d * HID + h * DHEAD);
    const float4* pa = reinterpret_cast<const float4*>(attn + h * DHEAD);
    float acc = 0.f;
    #pragma unroll
    for (int i = 0; i < DHEAD / 4; ++i) {
        float4 a = ps[i], b = pd[i], w = pa[i];
        float x0 = a.x + b.x, x1 = a.y + b.y, x2 = a.z + b.z, x3 = a.w + b.w;
        x0 = x0 > 0.f ? x0 : NEG_SLOPE * x0;
        x1 = x1 > 0.f ? x1 : NEG_SLOPE * x1;
        x2 = x2 > 0.f ? x2 : NEG_SLOPE * x2;
        x3 = x3 > 0.f ? x3 : NEG_SLOPE * x3;
        acc += x0 * w.x + x1 * w.y + x2 * w.z + x3 * w.w;
    }
    e[t] = acc;
}

__global__ __launch_bounds__(256)
void edge_segmax(const float* __restrict__ e, const int* __restrict__ dst,
                 unsigned int* __restrict__ mkeys)
{
    int t = blockIdx.x * blockDim.x + threadIdx.x;
    if (t >= E_EDGES * HEADS) return;
    int d = dst[t >> 3], h = t & 7;
    atomicMax(&mkeys[d * HEADS + h], fmap(e[t]));
}

__global__ __launch_bounds__(256)
void edge_expsum(float* __restrict__ e, const int* __restrict__ dst,
                 const unsigned int* __restrict__ mkeys, float* __restrict__ denom)
{
    int t = blockIdx.x * blockDim.x + threadIdx.x;
    if (t >= E_EDGES * HEADS) return;
    int d = dst[t >> 3], h = t & 7;
    float m  = funmap(mkeys[d * HEADS + h]);
    float ex = expf(e[t] - m);
    e[t] = ex;
    atomicAdd(&denom[d * HEADS + h], ex);
}

__global__ __launch_bounds__(256)
void edge_aggregate(const float* __restrict__ ex, const float* __restrict__ denom,
                    const float* __restrict__ hs, const int* __restrict__ src,
                    const int* __restrict__ dst, float* __restrict__ rst)
{
    int t = blockIdx.x * blockDim.x + threadIdx.x;
    if (t >= E_EDGES * HEADS) return;
    int eidx = t >> 3, h = t & 7;
    int s = src[eidx], d = dst[eidx];
    float a = ex[t] / denom[d * HEADS + h];
    const float4* ps = reinterpret_cast<const float4*>(hs + (size_t)s * HID + h * DHEAD);
    float* pr = rst + (size_t)d * HID + h * DHEAD;
    #pragma unroll
    for (int i = 0; i < DHEAD / 4; ++i) {
        float4 v = ps[i];
        atomicAdd(pr + i * 4 + 0, a * v.x);
        atomicAdd(pr + i * 4 + 1, a * v.y);
        atomicAdd(pr + i * 4 + 2, a * v.z);
        atomicAdd(pr + i * 4 + 3, a * v.w);
    }
}

// h = relu(rst + h); also refresh the f16 mirror used as the next GEMM A.
__global__ __launch_bounds__(256)
void residual_relu(float* __restrict__ h, _Float16* __restrict__ h16,
                   const float* __restrict__ rst, int count)
{
    int t = blockIdx.x * blockDim.x + threadIdx.x;
    if (t >= count) return;
    float v = rst[t] + h[t];
    v = v > 0.f ? v : 0.f;
    h[t] = v;
    h16[t] = (_Float16)v;
}

__global__ __launch_bounds__(256)
void graph_max(const float* __restrict__ h, const int* __restrict__ gid,
               unsigned int* __restrict__ gkeys)
{
    int t = blockIdx.x * blockDim.x + threadIdx.x;
    if (t >= N_NODES * HID) return;
    int node = t >> 8;
    atomicMax(&gkeys[gid[node] * HID + (t & 255)], fmap(h[t]));
}

__global__ __launch_bounds__(256)
void graph_max_finalize(const unsigned int* __restrict__ gkeys,
                        _Float16* __restrict__ hg16)
{
    int t = blockIdx.x * blockDim.x + threadIdx.x;
    if (t >= G_GRAPHS * HID) return;
    unsigned int k = gkeys[t];
    hg16[t] = (_Float16)((k == 0u) ? 0.f : funmap(k));   // empty graph -> 0
}

// ---------------------------------------------------------------------------
static inline void launch_gemm(const _Float16* A, const _Float16* BT, const float* bias,
                               float* C, _Float16* C16, int M, int N, int K, int relu,
                               hipStream_t s)
{
    dim3 grid((N + BN - 1) / BN, (M + BM - 1) / BM);
    gemm_bias_wmma<<<grid, 256, 0, s>>>(A, BT, bias, C, C16, M, N, K, relu);
}

extern "C" void kernel_launch(void* const* d_in, const int* in_sizes, int n_in,
                              void* d_out, int out_size, void* d_ws, size_t ws_size,
                              hipStream_t stream)
{
    (void)in_sizes; (void)n_in; (void)out_size; (void)ws_size;

    const float* x    = (const float*)d_in[0];
    const float* Win  = (const float*)d_in[1];
    const float* b_in = (const float*)d_in[2];
    const float* Wsrc = (const float*)d_in[3];
    const float* bsrc = (const float*)d_in[4];
    const float* Wdst = (const float*)d_in[5];
    const float* bdst = (const float*)d_in[6];
    const float* attn = (const float*)d_in[7];
    const float* Wc1  = (const float*)d_in[8];
    const float* bc1  = (const float*)d_in[9];
    const float* Wc2  = (const float*)d_in[10];
    const float* bc2  = (const float*)d_in[11];
    const float* Wc3  = (const float*)d_in[12];
    const float* bc3  = (const float*)d_in[13];
    const int*   src  = (const int*)d_in[14];
    const int*   dst  = (const int*)d_in[15];
    const int*   gid  = (const int*)d_in[16];
    float* out = (float*)d_out;

    // ---- workspace layout ----
    char* w = (char*)d_ws;
    size_t o = 0;
    auto carve = [&](size_t bytes) -> char* {
        char* p = w + o;
        o += (bytes + 255) & ~(size_t)255;
        return p;
    };
    float*        h     = (float*)       carve((size_t)N_NODES * HID * 4);
    float*        hs    = (float*)       carve((size_t)N_NODES * HID * 4);
    float*        hd    = (float*)       carve((size_t)N_NODES * HID * 4);  // reused as rst
    float*        e     = (float*)       carve((size_t)E_EDGES * HEADS * 4);
    unsigned int* mkeys = (unsigned int*)carve((size_t)N_NODES * HEADS * 4); // reused as gkeys
    float*        denom = (float*)       carve((size_t)N_NODES * HEADS * 4);
    _Float16*     x16   = (_Float16*)    carve((size_t)N_NODES * IN_DIM * 2);
    _Float16*     h16   = (_Float16*)    carve((size_t)N_NODES * HID * 2);
    _Float16*     WinT  = (_Float16*)    carve((size_t)IN_DIM * HID * 2);
    _Float16*     WsrcT = (_Float16*)    carve((size_t)3 * HID * HID * 2);
    _Float16*     WdstT = (_Float16*)    carve((size_t)3 * HID * HID * 2);
    _Float16*     Wc1T  = (_Float16*)    carve((size_t)HID * HID * 2);
    _Float16*     Wc2T  = (_Float16*)    carve((size_t)HID * (HID / 2) * 2);
    _Float16*     Wc3T  = (_Float16*)    carve((size_t)(HID / 2) * OUT_DIM * 2);
    _Float16*     hg16  = (_Float16*)    carve((size_t)G_GRAPHS * HID * 2);
    float*        z1    = (float*)       carve((size_t)G_GRAPHS * HID * 4);
    _Float16*     z116  = (_Float16*)    carve((size_t)G_GRAPHS * HID * 2);
    float*        z2    = (float*)       carve((size_t)G_GRAPHS * (HID / 2) * 4);
    _Float16*     z216  = (_Float16*)    carve((size_t)G_GRAPHS * (HID / 2) * 2);

    const int EH  = E_EDGES * HEADS;
    const int NF  = N_NODES * HID;
    const int ebl = (EH + 255) / 256;
    const int nbl = (NF + 255) / 256;
    auto blks = [](int n) { return (n + 255) / 256; };

    // ---- one-shot f16 conversions: A copies + transposed weights ----
    cvt_f32_f16<<<blks(N_NODES * IN_DIM), 256, 0, stream>>>(x, x16, N_NODES * IN_DIM);
    cvt_transpose_f32_f16<<<blks(IN_DIM * HID), 256, 0, stream>>>(Win, WinT, IN_DIM, HID);
    for (int l = 0; l < 3; ++l) {
        cvt_transpose_f32_f16<<<blks(HID * HID), 256, 0, stream>>>(
            Wsrc + (size_t)l * HID * HID, WsrcT + (size_t)l * HID * HID, HID, HID);
        cvt_transpose_f32_f16<<<blks(HID * HID), 256, 0, stream>>>(
            Wdst + (size_t)l * HID * HID, WdstT + (size_t)l * HID * HID, HID, HID);
    }
    cvt_transpose_f32_f16<<<blks(HID * HID), 256, 0, stream>>>(Wc1, Wc1T, HID, HID);
    cvt_transpose_f32_f16<<<blks(HID * (HID / 2)), 256, 0, stream>>>(Wc2, Wc2T, HID, HID / 2);
    cvt_transpose_f32_f16<<<blks((HID / 2) * OUT_DIM), 256, 0, stream>>>(Wc3, Wc3T, HID / 2, OUT_DIM);

    // ---- input projection (writes both f32 h and f16 mirror) ----
    launch_gemm(x16, WinT, b_in, h, h16, N_NODES, HID, IN_DIM, 0, stream);

    // ---- 3 GATv2 layers ----
    for (int l = 0; l < 3; ++l) {
        launch_gemm(h16, WsrcT + (size_t)l * HID * HID, bsrc + (size_t)l * HID,
                    hs, nullptr, N_NODES, HID, HID, 0, stream);
        launch_gemm(h16, WdstT + (size_t)l * HID * HID, bdst + (size_t)l * HID,
                    hd, nullptr, N_NODES, HID, HID, 0, stream);

        edge_scores<<<ebl, 256, 0, stream>>>(hs, hd, attn + (size_t)l * HEADS * DHEAD,
                                             src, dst, e);
        hipMemsetAsync(mkeys, 0, (size_t)N_NODES * HEADS * 4, stream);
        edge_segmax<<<ebl, 256, 0, stream>>>(e, dst, mkeys);
        hipMemsetAsync(denom, 0, (size_t)N_NODES * HEADS * 4, stream);
        edge_expsum<<<ebl, 256, 0, stream>>>(e, dst, mkeys, denom);

        hipMemsetAsync(hd, 0, (size_t)N_NODES * HID * 4, stream);   // hd := rst
        edge_aggregate<<<ebl, 256, 0, stream>>>(e, denom, hs, src, dst, hd);
        residual_relu<<<nbl, 256, 0, stream>>>(h, h16, hd, NF);
    }

    // ---- per-graph max readout ----
    hipMemsetAsync(mkeys, 0, (size_t)G_GRAPHS * HID * 4, stream);   // mkeys := gkeys
    graph_max<<<nbl, 256, 0, stream>>>(h, gid, mkeys);
    graph_max_finalize<<<blks(G_GRAPHS * HID), 256, 0, stream>>>(mkeys, hg16);

    // ---- classifier ----
    launch_gemm(hg16, Wc1T, bc1, z1, z116, G_GRAPHS, HID,     HID,     1, stream);
    launch_gemm(z116, Wc2T, bc2, z2, z216, G_GRAPHS, HID / 2, HID,     1, stream);
    launch_gemm(z216, Wc3T, bc3, out, nullptr, G_GRAPHS, OUT_DIM, HID / 2, 0, stream);
}
